// MMCL_PGD_76802605187498
// MI455X (gfx1250) — matrix operand order repaired and checked by reading, not compiled, not run on hardware
//
#include <hip/hip_runtime.h>
#include <hip/hip_bf16.h>

typedef __attribute__((ext_vector_type(16))) _Float16 v16h;
typedef __attribute__((ext_vector_type(8)))  _Float16 v8h;
typedef __attribute__((ext_vector_type(8)))  float    v8f;

#define BSZ   256
#define NN    255
#define DIM   128
#define GAMMA 0.07f
#define REGC  0.1f
#define ITERS 1000
#define ETA   0.001f

// ---------------------------------------------------------------------------
// k1: normalize z -> ftr (512 x 128), row squared-norms xn (512)
// z layout (256,2,128); norm over the 2-axis per (b,d).
// ---------------------------------------------------------------------------
__global__ __launch_bounds__(128) void ftr_kernel(const float* __restrict__ z,
                                                  float* __restrict__ ftr,
                                                  float* __restrict__ xn) {
    const int b = blockIdx.x;     // 0..255
    const int d = threadIdx.x;    // 0..127
    float z0 = z[b * 256 + d];
    float z1 = z[b * 256 + 128 + d];
    float nrm = fmaxf(sqrtf(z0 * z0 + z1 * z1), 1e-12f);
    float f0 = z0 / nrm, f1 = z1 / nrm;
    ftr[b * DIM + d]           = f0;
    ftr[(b + BSZ) * DIM + d]   = f1;
    __shared__ float s0[128], s1[128];
    s0[d] = f0 * f0; s1[d] = f1 * f1;
    __syncthreads();
    for (int st = 64; st > 0; st >>= 1) {
        if (d < st) { s0[d] += s0[d + st]; s1[d] += s1[d + st]; }
        __syncthreads();
    }
    if (d == 0) { xn[b] = s0[0]; xn[b + BSZ] = s1[0]; }
}

// ---------------------------------------------------------------------------
// k2: K[i,j] = exp(-gamma*(xn_i + xn_j - 2 ftr_i.ftr_j)), i<256, j<512.
//  j<256  -> KKh (f16, row-major 256x256)        [feeds WMMA GEMM]
//  j>=256 -> KS[i*256 + (j-256)] (f32)           [feeds loss; KS[k][b]=K[k,256+b]]
// ---------------------------------------------------------------------------
__global__ __launch_bounds__(256) void kmat_kernel(const float* __restrict__ ftr,
                                                   const float* __restrict__ xn,
                                                   _Float16* __restrict__ KKh,
                                                   float* __restrict__ KS) {
    const int i = blockIdx.x;
    const int t = threadIdx.x;
    __shared__ float fi[DIM];
    if (t < DIM) fi[t] = ftr[i * DIM + t];
    __syncthreads();
    const float xni = xn[i];
    #pragma unroll
    for (int rep = 0; rep < 2; ++rep) {
        const int j = t + rep * 256;
        const float* fj = ftr + j * DIM;
        float dot = 0.f;
        #pragma unroll 4
        for (int d = 0; d < DIM; ++d) dot += fi[d] * fj[d];
        float Kv = expf(-GAMMA * (xni + xn[j] - 2.0f * dot));
        if (j < BSZ) KKh[i * BSZ + j] = (_Float16)Kv;
        else         KS[i * BSZ + (j - BSZ)] = Kv;
    }
}

// ---------------------------------------------------------------------------
// k3: persistent PGD. 16 blocks x 512 threads (16 waves).
//  Wave w: (a) GEMM M-tile rows [16w,16w+16) of G = KK @ Yhat  (A held in VGPRs)
//          (b) owner of batch column b = b0 + w (expanded alpha in registers)
// ---------------------------------------------------------------------------
__global__ __launch_bounds__(512) void pgd_kernel(const float* __restrict__ alpha_init,
                                                  const _Float16* __restrict__ KKh,
                                                  const float* __restrict__ KS,
                                                  float* __restrict__ parts) {
    const int wave = threadIdx.x >> 5;
    const int lane = threadIdx.x & 31;
    const int b0   = blockIdx.x << 4;
    const int bgc  = b0 + wave;                 // global batch column for this wave

    __shared__ __attribute__((aligned(32))) _Float16 yh[16][264]; // Yhat, col-major, padded
    __shared__ __attribute__((aligned(32))) float    Gs[16][260]; // G,  col-major, padded
    __shared__ float part[16][8];

    // ---- A operand (loop-invariant): rows m0..m0+15 of KK, all K, in VGPRs.
    // ISA A 16x32 f16 layout: lane L -> row M = m0 + (L&15);
    // lanes<16 hold K {k0+0..7, k0+16..23}; lanes>=16 hold K {k0+8..15, k0+24..31}.
    const int m   = (wave << 4) + (lane & 15);
    const int off = (lane < 16) ? 0 : 8;
    v16h Areg[8];
    #pragma unroll
    for (int kb = 0; kb < 8; ++kb) {
        const _Float16* p = KKh + m * BSZ + kb * 32 + off;
        union { v16h v; v8h h[2]; } u;
        u.h[0] = *(const v8h*)(p);
        u.h[1] = *(const v8h*)(p + 16);
        Areg[kb] = u.v;
    }

    // ---- expanded alpha (lane owns rows k = 8*lane .. 8*lane+7 of column bgc)
    float a[8], ap[8], rowk[8];
    #pragma unroll
    for (int j = 0; j < 8; ++j) {
        const int k = lane * 8 + j;
        float v = 0.f;
        if (k != bgc) {
            const int i = k - (k > bgc ? 1 : 0);
            v = alpha_init[bgc * NN + i];
            v = fminf(fmaxf(v, 0.f), 1.f);        // clip(relu(.),0,C)
        }
        a[j] = v; ap[j] = v;
        rowk[j] = (float)KKh[bgc * BSZ + k];       // KK[bgc, k]
    }

    const int n    = lane & 15;
    const int koff = (lane < 16) ? 0 : 8;
    const int mb   = (wave << 4) + ((lane < 16) ? 0 : 8);

    float y[8];
    for (int t = 0; t < ITERS; ++t) {
        const float beta = (float)t / ((float)t + 3.0f);
        float sp = 0.f;
        #pragma unroll
        for (int j = 0; j < 8; ++j) {
            y[j] = a[j] + beta * (a[j] - ap[j]);   // y[bgc] stays 0 by invariant
            sp += y[j];
            yh[wave][lane * 8 + j] = (_Float16)y[j];
        }
        __syncthreads();

        // ---- GEMM: G[:, b0..b0+15] tile rows [16w,16w+16), K = 256 in 8 steps.
        // Preload ALL B tiles first (16 ds_load_b128 in flight, one wait),
        // then issue the 8 WMMAs back-to-back with no interleaved memory waits.
        // ISA B 32x16 f16 layout: lane L -> col N = L&15;
        // lanes<16 hold K {k0+0..7, k0+16..23}; lanes>=16 hold K {k0+8..15, k0+24..31}.
        v16h Breg[8];
        #pragma unroll
        for (int kb = 0; kb < 8; ++kb) {
            const _Float16* q = &yh[n][kb * 32 + koff];
            union { v16h v; v8h h[2]; } ub;
            ub.h[0] = *(const v8h*)(q);
            ub.h[1] = *(const v8h*)(q + 16);
            Breg[kb] = ub.v;
        }
        v8f acc = {};
        #pragma unroll
        for (int kb = 0; kb < 8; ++kb) {
            acc = __builtin_amdgcn_wmma_f32_16x16x32_f16(
                false, Areg[kb], false, Breg[kb], (short)0, acc, false, false);
        }
        // D layout: VGPR v -> M = m0 + v (lanes<16) / m0 + 8 + v (lanes>=16)
        #pragma unroll
        for (int v = 0; v < 8; ++v) Gs[n][mb + v] = acc[v];
        __syncthreads();

        // ---- grad + Nesterov update for column bgc (one wave per column)
        float S = sp;
        #pragma unroll
        for (int d = 16; d >= 1; d >>= 1) S += __shfl_xor(S, d, 32);
        const float Gb = Gs[wave][bgc];            // (KK Yhat)[bgc, b] == row . y
        float g[8]; float n2 = 0.f;
        #pragma unroll
        for (int j = 0; j < 8; ++j) {
            const int k = lane * 8 + j;
            float gv = Gs[wave][k] + S * (1.0f - rowk[j]) + REGC * y[j] - Gb - 2.0f;
            if (k == bgc) gv = 0.f;                // masked coordinate
            g[j] = gv; n2 += gv * gv;
        }
        #pragma unroll
        for (int d = 16; d >= 1; d >>= 1) n2 += __shfl_xor(n2, d, 32);
        const float inv = ETA / (sqrtf(n2) + 1e-12f);
        #pragma unroll
        for (int j = 0; j < 8; ++j) {
            float an = y[j] - inv * g[j];
            an = fminf(fmaxf(an, 0.f), 1.f);
            ap[j] = a[j]; a[j] = an;
        }
        // next iteration's first __syncthreads orders yh/Gs rewrites vs these reads
    }

    // ---- per-workgroup loss partials (fixed order -> deterministic)
    float ax = 0.f, neg = 0.f, csum = 0.f, c1 = 0.f, c6 = 0.f, c7 = 0.f;
    #pragma unroll
    for (int j = 0; j < 8; ++j) {
        const int k = lane * 8 + j;
        const float av = a[j];
        const float ks = KS[k * BSZ + bgc];        // K[k, 256+bgc]
        ax += av; neg += av * ks; csum += ks;
        c1 += (av == 1.0f) ? 1.f : 0.f;
        c6 += (av > 0.0f)  ? 1.f : 0.f;
        c7 += (av == 0.0f) ? 1.f : 0.f;
    }
    #pragma unroll
    for (int d = 16; d >= 1; d >>= 1) {
        ax += __shfl_xor(ax, d, 32);  neg  += __shfl_xor(neg, d, 32);
        csum += __shfl_xor(csum, d, 32);
        c1 += __shfl_xor(c1, d, 32);  c6   += __shfl_xor(c6, d, 32);
        c7 += __shfl_xor(c7, d, 32);
    }
    if (lane == 0) {
        const float pos = KS[bgc * BSZ + bgc];
        part[wave][0] = ax * pos;  part[wave][1] = neg;  part[wave][2] = pos;
        part[wave][3] = csum;      part[wave][4] = c1;   part[wave][5] = c6;
        part[wave][6] = c7;        part[wave][7] = 0.f;
    }
    __syncthreads();
    if (threadIdx.x < 8) {
        float s = 0.f;
        for (int c = 0; c < 16; ++c) s += part[c][threadIdx.x];
        parts[blockIdx.x * 8 + threadIdx.x] = s;
    }
}

// ---------------------------------------------------------------------------
// k4: combine 16x8 partials -> 6 output scalars
// ---------------------------------------------------------------------------
__global__ void finalize_kernel(const float* __restrict__ parts, float* __restrict__ out) {
    if (threadIdx.x == 0 && blockIdx.x == 0) {
        float P[8];
        for (int q = 0; q < 8; ++q) P[q] = 0.f;
        for (int w = 0; w < 16; ++w)
            for (int q = 0; q < 8; ++q) P[q] += parts[w * 8 + q];
        const float pos_loss = P[0] / 256.0f;
        const float neg_loss = P[1] / 256.0f;
        out[0] = neg_loss - pos_loss;                         // loss
        out[1] = P[2] / 256.0f;                               // pos.mean()
        out[2] = (P[3] - P[2]) / (256.0f * 255.0f);           // KnT.mean()
        out[3] = P[4] / (P[5] + 1e-10f);                      // sparsity
        out[4] = (P[6] - 256.0f) / (256.0f * 255.0f);         // num_zero (drop forced diag zeros)
        out[5] = 0.0f;
    }
}

// ---------------------------------------------------------------------------
extern "C" void kernel_launch(void* const* d_in, const int* in_sizes, int n_in,
                              void* d_out, int out_size, void* d_ws, size_t ws_size,
                              hipStream_t stream) {
    const float* z          = (const float*)d_in[0];   // 256*2*128
    const float* alpha_init = (const float*)d_in[1];   // 256*255
    float* out = (float*)d_out;                        // 6 f32

    // workspace layout (floats unless noted)
    float* wsf   = (float*)d_ws;
    float* ftr   = wsf;                 // 65536
    float* xn    = wsf + 65536;         // 512
    float* KS    = wsf + 66048;         // 65536
    float* parts = wsf + 131584;        // 128
    _Float16* KKh = (_Float16*)(wsf + 131712); // 65536 halves (128 KB)

    ftr_kernel<<<256, 128, 0, stream>>>(z, ftr, xn);
    kmat_kernel<<<256, 256, 0, stream>>>(ftr, xn, KKh, KS);
    pgd_kernel<<<16, 512, 0, stream>>>(alpha_init, KKh, KS, parts);
    finalize_kernel<<<1, 32, 0, stream>>>(parts, out);
}
